// BertSelfAttention_15513421873478
// MI455X (gfx1250) — matrix operand order repaired and checked by reading, not compiled, not run on hardware
//
#include <hip/hip_runtime.h>
#include <hip/hip_bf16.h>
#include <math.h>

#define D_MODEL 1024
#define N_H     16
#define HD      64
#define BATCH   8
#define SEQ     1024

typedef __attribute__((ext_vector_type(16))) __bf16       v16bf;
typedef __attribute__((ext_vector_type(8)))  float        v8f;
typedef __attribute__((ext_vector_type(4)))  unsigned int v4u;
typedef __attribute__((ext_vector_type(4)))  unsigned int u32x4;
typedef __attribute__((ext_vector_type(8)))  int          i32x8;
typedef __attribute__((ext_vector_type(4)))  int          i32x4;

union Frag16 { v16bf bf; v4u u4[2]; };
union Pack8  { v4u u; __bf16 e[8]; };

#if defined(__has_builtin)
#  if __has_builtin(__builtin_amdgcn_tensor_load_to_lds) && __has_builtin(__builtin_amdgcn_s_wait_tensorcnt)
#    define USE_TDM 1
#  endif
#endif
#ifndef USE_TDM
#  define USE_TDM 0
#endif

#if USE_TDM
// ---------------------------------------------------------------------------
// 2D Tensor-DMA load: global (row-major, bf16) -> LDS, optional row padding.
// D# packing per CDNA5 ISA §8.3/§8.4 (group0 128b, group1 256b), groups 2/3
// zero (2D tile: tile_dim2/3/4 = 0 = unused).  Issue once per workgroup from
// a single wave (TDM ignores EXEC; scalar branch prevents redundant issue).
// 6-arg builtin form (clang-23 / therock-10.0 headers).
// ---------------------------------------------------------------------------
__device__ __forceinline__ void tdm_load_2d_bf16(
    const void* gaddr, unsigned lds_byte_addr,
    unsigned tensor_d0, unsigned tensor_d1,         // elements
    unsigned tile_d0, unsigned tile_d1,             // elements
    unsigned long long stride0_elems,
    int pad_en, int pad_interval_code, int pad_amount_code)
{
    unsigned long long ga = (unsigned long long)gaddr;
    u32x4 g0;
    g0[0] = 1u;                                     // count=1, user mode
    g0[1] = lds_byte_addr;                          // lds_addr [63:32]
    g0[2] = (unsigned)(ga & 0xFFFFFFFFu);           // global_addr [95:64]
    g0[3] = (unsigned)((ga >> 32) & 0x01FFFFFFu)    // global_addr [120:96]
          | (2u << 30);                             // type=2 ("image") [127:126]
    i32x8 g1;
    g1[0] = (int)((1u << 16)                        // data_size=1 (2 bytes)
          | ((unsigned)pad_en << 20)
          | ((unsigned)pad_interval_code << 22)
          | ((unsigned)pad_amount_code << 25));     // wg_mask=0 (not in cluster)
    g1[1] = (int)((tensor_d0 & 0xFFFFu) << 16);     // tensor_dim0[15:0] @ [63:48]
    g1[2] = (int)((tensor_d0 >> 16) | ((tensor_d1 & 0xFFFFu) << 16));
    g1[3] = (int)((tensor_d1 >> 16) | ((tile_d0 & 0xFFFFu) << 16)); // tile_dim0
    g1[4] = (int)(tile_d1 & 0xFFFFu);               // tile_dim1; tile_dim2=0
    g1[5] = (int)(unsigned)(stride0_elems & 0xFFFFFFFFu);  // dim0_stride[31:0]
    g1[6] = (int)(unsigned)((stride0_elems >> 32) & 0xFFFFu); // [47:32]; dim1_stride=0
    g1[7] = 0;
    i32x4 z4 = {0, 0, 0, 0};
    i32x8 z8 = {0, 0, 0, 0, 0, 0, 0, 0};
    __builtin_amdgcn_tensor_load_to_lds(g0, g1, z4, z4, z8, 0);
}
#endif

// ---------------------------------------------------------------------------
// fp32 -> bf16 conversion, 4 elements / thread
// ---------------------------------------------------------------------------
__global__ void cvt_bf16(const float* __restrict__ in, __bf16* __restrict__ out, int n4) {
    int i = blockIdx.x * 256 + threadIdx.x;
    if (i >= n4) return;
    float4 f = ((const float4*)in)[i];
    union { __bf16 e[4]; unsigned long long u; } r;
    r.e[0] = (__bf16)f.x; r.e[1] = (__bf16)f.y;
    r.e[2] = (__bf16)f.z; r.e[3] = (__bf16)f.w;
    ((unsigned long long*)out)[i] = r.u;
}

// ---------------------------------------------------------------------------
// bf16 GEMM: C[M,N] = A[M,K] @ B[K,N] + bias
// MODE 0: scatter bf16 into Q/K/V [B,H,S,hd];  MODE 1: f32 out row-major
// Block tile 128x128x32, 256 threads (8 waves), wave tile 32x64 (2x4 WMMA).
// A tile loaded by the Tensor Data Mover (async, TENSORcnt, LDS row padding
// 64B+16B -> 80B stride); B tile transposed-stored manually.
// ---------------------------------------------------------------------------
template<int MODE>
__global__ __launch_bounds__(256)
void gemm_bf16(const __bf16* __restrict__ A, const __bf16* __restrict__ Bw,
               const float* __restrict__ bias,
               __bf16* __restrict__ Qo, __bf16* __restrict__ Ko, __bf16* __restrict__ Vo,
               float* __restrict__ Fo, int M, int N, int K)
{
    constexpr int BM = 128, BN = 128, BK = 32, LS = 40; // LS: LDS row stride (bf16) = 80B
    __shared__ __bf16 As[BM * LS];
    __shared__ __bf16 Bs[BN * LS];   // stored transposed: Bs[n][k]

    const int tid  = threadIdx.x;
    const int lane = tid & 31, wave = tid >> 5;
    const int n16  = lane & 15, hl = lane >> 4;
    const int wr   = wave >> 1, wc = wave & 1;     // 4x2 wave grid
    const int bm   = blockIdx.y * BM, bn = blockIdx.x * BN;

    const int ar = tid >> 2, ac = (tid & 3) << 3;        // A loader (fallback)
    const int bk = tid >> 4, bnn = (tid & 15) << 3;      // B loader: 2 k-rows x 8 cols

    v8f acc[2][4] = {};

    for (int kk = 0; kk < K; kk += BK) {
#if USE_TDM
        if (wave == 0) {
            // 128 rows x 32 bf16: row = 64B = 16 DW -> pad_interval code 3 (16 DW),
            // pad 16B = 4 DW -> pad_amount code 3.
            tdm_load_2d_bf16(A + (size_t)bm * K + kk, (unsigned)(size_t)(void*)As,
                             (unsigned)K, (unsigned)M, BK, BM,
                             (unsigned long long)K, 1, 3, 3);
        }
#else
        #pragma unroll
        for (int rr = 0; rr < 2; ++rr) {
            int r = ar + rr * 64;
            v4u t = *(const v4u*)(A + (size_t)(bm + r) * K + kk + ac);
            *(v4u*)(As + r * LS + ac) = t;
        }
#endif
        #pragma unroll
        for (int rr = 0; rr < 2; ++rr) {
            int kr = bk + rr * 16;
            Pack8 p; p.u = *(const v4u*)(Bw + (size_t)(kk + kr) * N + bn + bnn);
            #pragma unroll
            for (int i = 0; i < 8; ++i) Bs[(bnn + i) * LS + kr] = p.e[i];
        }
        if (kk + BK < K)   // prefetch next B tile rows (global_prefetch_b8)
            __builtin_prefetch((const void*)(Bw + (size_t)(kk + BK + bk) * N + bn + bnn), 0, 3);
#if USE_TDM
        if (wave == 0) __builtin_amdgcn_s_wait_tensorcnt(0);
#endif
        __syncthreads();

        Frag16 af[2], bfr[4];
        #pragma unroll
        for (int i = 0; i < 2; ++i) {   // A frag: row m=n16, k in VGPR-pattern order
            const char* p = (const char*)(As + (wr * 32 + i * 16 + n16) * LS);
            af[i].u4[0] = *(const v4u*)(p + hl * 16);        // k = hl*8 .. +7
            af[i].u4[1] = *(const v4u*)(p + 32 + hl * 16);   // k = 16 + hl*8 ..
        }
        #pragma unroll
        for (int j = 0; j < 4; ++j) {   // B frag: col n=n16, k in VGPR-pattern order
            const char* p = (const char*)(Bs + (wc * 64 + j * 16 + n16) * LS);
            bfr[j].u4[0] = *(const v4u*)(p + hl * 16);
            bfr[j].u4[1] = *(const v4u*)(p + 32 + hl * 16);
        }
        #pragma unroll
        for (int i = 0; i < 2; ++i)
            #pragma unroll
            for (int j = 0; j < 4; ++j)
                acc[i][j] = __builtin_amdgcn_wmma_f32_16x16x32_bf16(
                    false, af[i].bf, false, bfr[j].bf, (short)0, acc[i][j], false, false);
        __syncthreads();
    }

    #pragma unroll
    for (int i = 0; i < 2; ++i)
        #pragma unroll
        for (int j = 0; j < 4; ++j)
            #pragma unroll
            for (int e = 0; e < 8; ++e) {
                int gm = bm + wr * 32 + i * 16 + e + 8 * hl;   // C layout: M=e+8*half
                int gn = bn + wc * 64 + j * 16 + n16;          // N = lane%16
                float val = acc[i][j][e] + bias[gn];
                if (MODE == 0) {
                    int which = gn >> 10, d2 = gn & 1023;
                    int h = d2 >> 6, hd = d2 & 63;
                    int b = gm >> 10, s = gm & 1023;
                    size_t off = ((size_t)(b * N_H + h) * SEQ + s) * HD + hd;
                    __bf16* dst = which == 0 ? Qo : (which == 1 ? Ko : Vo);
                    dst[off] = (__bf16)val;
                } else {
                    Fo[(size_t)gm * N + gn] = val;
                }
            }
}

// ---------------------------------------------------------------------------
// Flash attention: 1 WG = 128 q rows of one (b,h); 8 waves x 16 rows.
// K chunk (64x64) loaded by TDM (row = 128B = 32 DW, pad 32B = 8 DW -> 160B
// stride); V transposed manually; online softmax; PV via per-wave LDS P.
// ---------------------------------------------------------------------------
__global__ __launch_bounds__(256)
void flash_attn(const __bf16* __restrict__ Q, const __bf16* __restrict__ Kg,
                const __bf16* __restrict__ V, const int* __restrict__ mask,
                __bf16* __restrict__ O)
{
    constexpr int LS = 80;                    // 160B row stride
    __shared__ __bf16 Ks[64 * LS];            // [key][hd]
    __shared__ __bf16 Vt[64 * LS];            // [hd][key] (transposed)
    __shared__ float  mbias[64];
    __shared__ __bf16 Pb[8][16 * LS];         // per-wave P bounce buffer

    const int tid  = threadIdx.x;
    const int lane = tid & 31, wave = tid >> 5;
    const int n16  = lane & 15, hl = lane >> 4;
    const int b = blockIdx.z, h = blockIdx.y;
    const size_t head = (size_t)(b * N_H + h) * SEQ * HD;
    const int qrow = blockIdx.x * 128 + wave * 16;

    // Q fragments (A-matrix, 16 rows x 64 hd = 2 frags), loaded once
    Frag16 qf[2];
    {
        const char* qp = (const char*)(Q + head + (size_t)(qrow + n16) * HD);
        #pragma unroll
        for (int f = 0; f < 2; ++f) {
            qf[f].u4[0] = *(const v4u*)(qp + f * 64 + hl * 16);
            qf[f].u4[1] = *(const v4u*)(qp + f * 64 + 32 + hl * 16);
        }
    }

    float rmax[8], rsum[8];
    #pragma unroll
    for (int e = 0; e < 8; ++e) { rmax[e] = -1e30f; rsum[e] = 0.f; }
    v8f oacc[4] = {};

    const int r0 = tid >> 3, c8 = (tid & 7) << 3;

    for (int kc = 0; kc < SEQ / 64; ++kc) {
#if USE_TDM
        if (wave == 0) {
            tdm_load_2d_bf16(Kg + head + (size_t)kc * 64 * HD, (unsigned)(size_t)(void*)Ks,
                             HD, 64, HD, 64, (unsigned long long)HD, 1, 4, 7);
        }
#endif
        #pragma unroll
        for (int rr = 0; rr < 2; ++rr) {
            int r = r0 + rr * 32;
#if !USE_TDM
            *(v4u*)(Ks + r * LS + c8) =
                *(const v4u*)(Kg + head + (size_t)(kc * 64 + r) * HD + c8);
#endif
            Pack8 p; p.u = *(const v4u*)(V + head + (size_t)(kc * 64 + r) * HD + c8);
            #pragma unroll
            for (int i = 0; i < 8; ++i) Vt[(c8 + i) * LS + r] = p.e[i];
        }
        if (tid < 64) mbias[tid] = mask[b * SEQ + kc * 64 + tid] ? 0.f : -1e9f;
#if USE_TDM
        if (wave == 0) __builtin_amdgcn_s_wait_tensorcnt(0);
#endif
        __syncthreads();

        // scores: 4 key-subtiles of 16, hd contraction = 2 WMMAs each
        float sc[4][8];
        #pragma unroll
        for (int t = 0; t < 4; ++t) {
            v8f s8 = {};
            #pragma unroll
            for (int f = 0; f < 2; ++f) {
                Frag16 kb;
                const char* p = (const char*)(Ks + (t * 16 + n16) * LS);
                kb.u4[0] = *(const v4u*)(p + f * 64 + hl * 16);
                kb.u4[1] = *(const v4u*)(p + f * 64 + 32 + hl * 16);
                s8 = __builtin_amdgcn_wmma_f32_16x16x32_bf16(
                    false, qf[f].bf, false, kb.bf, (short)0, s8, false, false);
            }
            float mb = mbias[t * 16 + n16];
            #pragma unroll
            for (int e = 0; e < 8; ++e) sc[t][e] = s8[e] * 0.125f + mb;
        }

        // online softmax (row stats live per lane-half: rows M = e + 8*hl)
        float mloc[8];
        #pragma unroll
        for (int e = 0; e < 8; ++e)
            mloc[e] = fmaxf(fmaxf(sc[0][e], sc[1][e]), fmaxf(sc[2][e], sc[3][e]));
        #pragma unroll
        for (int off = 8; off >= 1; off >>= 1)
            #pragma unroll
            for (int e = 0; e < 8; ++e)
                mloc[e] = fmaxf(mloc[e], __shfl_xor(mloc[e], off, 32));

        float alpha[8];
        #pragma unroll
        for (int e = 0; e < 8; ++e) {
            float nm = fmaxf(rmax[e], mloc[e]);
            alpha[e] = __expf(rmax[e] - nm);
            rmax[e]  = nm;
        }
        float lsum[8];
        #pragma unroll
        for (int e = 0; e < 8; ++e) lsum[e] = 0.f;
        #pragma unroll
        for (int t = 0; t < 4; ++t)
            #pragma unroll
            for (int e = 0; e < 8; ++e) {
                sc[t][e] = __expf(sc[t][e] - rmax[e]);
                lsum[e] += sc[t][e];
            }
        #pragma unroll
        for (int off = 8; off >= 1; off >>= 1)
            #pragma unroll
            for (int e = 0; e < 8; ++e) lsum[e] += __shfl_xor(lsum[e], off, 32);
        #pragma unroll
        for (int e = 0; e < 8; ++e) rsum[e] = rsum[e] * alpha[e] + lsum[e];
        #pragma unroll
        for (int c = 0; c < 4; ++c)
            #pragma unroll
            for (int e = 0; e < 8; ++e) oacc[c][e] *= alpha[e];

        // P (C layout) -> per-wave LDS -> A fragments
        __bf16* pb = Pb[wave];
        #pragma unroll
        for (int t = 0; t < 4; ++t)
            #pragma unroll
            for (int e = 0; e < 8; ++e)
                pb[(e + 8 * hl) * LS + t * 16 + n16] = (__bf16)sc[t][e];

        #pragma unroll
        for (int f = 0; f < 2; ++f) {           // key contraction halves
            Frag16 pa;
            const char* p = (const char*)(pb + n16 * LS);
            pa.u4[0] = *(const v4u*)(p + f * 64 + hl * 16);
            pa.u4[1] = *(const v4u*)(p + f * 64 + 32 + hl * 16);
            #pragma unroll
            for (int c = 0; c < 4; ++c) {       // hd output column tiles
                Frag16 vb;
                const char* vp = (const char*)(Vt + (c * 16 + n16) * LS);
                vb.u4[0] = *(const v4u*)(vp + f * 64 + hl * 16);
                vb.u4[1] = *(const v4u*)(vp + f * 64 + 32 + hl * 16);
                oacc[c] = __builtin_amdgcn_wmma_f32_16x16x32_bf16(
                    false, pa.bf, false, vb.bf, (short)0, oacc[c], false, false);
            }
        }
        __syncthreads();
    }

    // normalize + store bf16 into [B,S,D] (ready as GEMM A for projection)
    #pragma unroll
    for (int c = 0; c < 4; ++c)
        #pragma unroll
        for (int e = 0; e < 8; ++e) {
            float val = oacc[c][e] / rsum[e];
            int s   = qrow + e + 8 * hl;
            int col = h * HD + c * 16 + n16;
            O[((size_t)b * SEQ + s) * D_MODEL + col] = (__bf16)val;
        }
}

// ---------------------------------------------------------------------------
extern "C" void kernel_launch(void* const* d_in, const int* in_sizes, int n_in,
                              void* d_out, int out_size, void* d_ws, size_t ws_size,
                              hipStream_t stream) {
    const float* x    = (const float*)d_in[0];
    const int*   mask = (const int*)d_in[1];
    const float* Wqkv = (const float*)d_in[2];
    const float* bqkv = (const float*)d_in[3];
    const float* Wp   = (const float*)d_in[4];
    const float* bp   = (const float*)d_in[5];

    const size_t XN  = (size_t)BATCH * SEQ * D_MODEL;    // 8M elements
    const size_t WQN = (size_t)D_MODEL * 3 * D_MODEL;    // 3M
    const size_t WPN = (size_t)D_MODEL * D_MODEL;        // 1M

    char* ws = (char*)d_ws;
    __bf16* xb  = (__bf16*)ws; ws += XN * 2;
    __bf16* wqb = (__bf16*)ws; ws += WQN * 2;
    __bf16* wpb = (__bf16*)ws; ws += WPN * 2;
    __bf16* Qb  = (__bf16*)ws; ws += XN * 2;
    __bf16* Kb  = (__bf16*)ws; ws += XN * 2;
    __bf16* Vb  = (__bf16*)ws; ws += XN * 2;
    __bf16* Ob  = (__bf16*)ws; ws += XN * 2;   // ~88 MiB total

    cvt_bf16<<<(unsigned)(XN / 4 / 256), 256, 0, stream>>>(x, xb, (int)(XN / 4));
    cvt_bf16<<<(unsigned)(WQN / 4 / 256), 256, 0, stream>>>(Wqkv, wqb, (int)(WQN / 4));
    cvt_bf16<<<(unsigned)(WPN / 4 / 256), 256, 0, stream>>>(Wp, wpb, (int)(WPN / 4));

    gemm_bf16<0><<<dim3(3 * D_MODEL / 128, BATCH * SEQ / 128), 256, 0, stream>>>(
        xb, wqb, bqkv, Qb, Kb, Vb, nullptr, BATCH * SEQ, 3 * D_MODEL, D_MODEL);

    flash_attn<<<dim3(SEQ / 128, N_H, BATCH), 256, 0, stream>>>(Qb, Kb, Vb, mask, Ob);

    gemm_bf16<1><<<dim3(D_MODEL / 128, BATCH * SEQ / 128), 256, 0, stream>>>(
        Ob, wpb, bp, nullptr, nullptr, nullptr, (float*)d_out, BATCH * SEQ, D_MODEL, D_MODEL);
}
